// SwitchMLP_11793980194842
// MI455X (gfx1250) — matrix operand order repaired and checked by reading, not compiled, not run on hardware
//
#include <hip/hip_runtime.h>
#include <hip/hip_bf16.h>

// ---------------------------------------------------------------------------
// SwitchMLP for MI455X (gfx1250, wave32, WMMA bf16 + TDM staging)
//   h   = tanh(x @ W1 + b1)                    (bf16 staged, f32 accum WMMA)
//   h2  = tanh(h @ Wsw[sw[b]] + bsw[sw[b]])    (grouped GEMM via counting sort)
//   out = h2 @ W2 + b2                         (A tile staged via tensor_load_to_lds)
// ---------------------------------------------------------------------------

#define B_ROWS 4096
#define IN_DIM 1024
#define H_DIM  2048
#define OUT_DIM 1024
#define N_EXP  16
#define MAX_TILES 48

typedef __attribute__((ext_vector_type(16))) __bf16 v16bf;
typedef __attribute__((ext_vector_type(8)))  float  v8f;
typedef __attribute__((ext_vector_type(8)))  unsigned short v8us;
typedef __attribute__((ext_vector_type(4)))  float  f4v;
typedef __attribute__((ext_vector_type(4)))  unsigned int u4v;
typedef __attribute__((ext_vector_type(2)))  unsigned int u2v;
typedef __attribute__((ext_vector_type(4)))  unsigned int u32x4;
typedef __attribute__((ext_vector_type(8)))  int i32x8;
typedef __attribute__((ext_vector_type(4)))  int i32x4;

#if __has_builtin(__builtin_amdgcn_tensor_load_to_lds) && \
    __has_builtin(__builtin_amdgcn_s_wait_tensorcnt)
#define USE_TDM 1
#else
#define USE_TDM 0
#endif

__device__ __forceinline__ unsigned short f2bf(float f) {
  unsigned int u = __builtin_bit_cast(unsigned int, f);
  unsigned int r = u + 0x7FFFu + ((u >> 16) & 1u);   // round-to-nearest-even
  return (unsigned short)(r >> 16);
}

// Branch-free tanh: (1 - e^-2|x|) / (1 + e^-2|x|), sign restored.
// Uses v_exp_f32 / v_rcp_f32 directly -> ~7 VALU ops, no EXEC divergence.
__device__ __forceinline__ float fast_tanh(float x) {
  float ax = __builtin_fabsf(x);
  float t  = __builtin_amdgcn_exp2f(ax * -2.8853900817779268f);  // e^(-2ax)
  float y  = (1.0f - t) * __builtin_amdgcn_rcpf(1.0f + t);
  return __builtin_copysignf(y, x);
}

// ---------------------------------------------------------------------------
// Routing kernels: argmax -> counts -> offsets/tiles -> scatter perm
// ---------------------------------------------------------------------------
__global__ void init_kernel(int* counts, int* cursor, int* numTiles) {
  int t = threadIdx.x;
  if (t < N_EXP) { counts[t] = 0; cursor[t] = 0; }
  if (t == 0) *numTiles = 0;
}

__global__ void router_kernel(const int* __restrict__ logits,
                              int* __restrict__ sw, int* __restrict__ counts) {
  int b = blockIdx.x * blockDim.x + threadIdx.x;
  if (b >= B_ROWS) return;
  const int* l = logits + b * N_EXP;
  int best = l[0], bi = 0;
#pragma unroll
  for (int s = 1; s < N_EXP; ++s) {
    int v = l[s];
    if (v > best) { best = v; bi = s; }   // first occurrence wins (argmax)
  }
  sw[b] = bi;
  atomicAdd(&counts[bi], 1);
}

__global__ void scan_kernel(const int* __restrict__ counts,
                            int* __restrict__ offsets, int* __restrict__ cursor,
                            int* __restrict__ tileExpert, int* __restrict__ tileP0,
                            int* __restrict__ tilePend, int* __restrict__ numTiles) {
  if (threadIdx.x != 0 || blockIdx.x != 0) return;
  int off = 0, T = 0;
  for (int e = 0; e < N_EXP; ++e) {
    offsets[e] = off;
    cursor[e] = off;
    int end = off + counts[e];
    for (int p = off; p < end; p += 128) {
      tileExpert[T] = e; tileP0[T] = p; tilePend[T] = end; ++T;
    }
    off = end;
  }
  offsets[N_EXP] = off;
  *numTiles = T;
}

__global__ void scatter_kernel(const int* __restrict__ sw,
                               int* __restrict__ cursor, int* __restrict__ perm) {
  int b = blockIdx.x * blockDim.x + threadIdx.x;
  if (b >= B_ROWS) return;
  int pos = atomicAdd(&cursor[sw[b]], 1);
  perm[pos] = b;
}

// ---------------------------------------------------------------------------
// Tiled WMMA GEMM: C[M,N] = act(A[M,K] @ Bw[K,N] + bias)
//   ASRC: 0 = A is f32 global (convert), 1 = A is bf16 (ws buffer)
//   GATHER: rows gathered through perm[], B/bias selected per-tile expert
//   Dense bf16 A (ASRC==1 && !GATHER): staged with the Tensor Data Mover.
//   Block: 256 threads = 8 waves (4 in M x 2 in N); macro-tile 128x128, K-step 32
// ---------------------------------------------------------------------------
template<int ASRC, bool GATHER, bool DO_TANH, bool OUT_BF16>
__global__ __launch_bounds__(256)
void gemm_kernel(const float* __restrict__ Af32,
                 const unsigned short* __restrict__ Abf,
                 const float* __restrict__ Ball,
                 const float* __restrict__ biasAll,
                 float* __restrict__ Cf32,
                 unsigned short* __restrict__ Cbf,
                 int K, int lda, int ldb, int ldc,
                 const int* __restrict__ perm,
                 const int* __restrict__ tileExpert,
                 const int* __restrict__ tileP0,
                 const int* __restrict__ tilePend,
                 const int* __restrict__ numTiles,
                 int biasStride) {
  // LDS tiles: A row-major [m][k], B transposed [n][k]; stride 40 halfwords
  // (64B data + 16B pad per row) keeps ds_load_b128 fragments 16B-aligned and
  // spreads banks. The TDM pad fields reproduce this layout exactly.
  __shared__ __attribute__((aligned(16))) unsigned short As[128 * 40];
  __shared__ __attribute__((aligned(16))) unsigned short Bs[128 * 40];
  __shared__ int permS[128];

  const int tid  = threadIdx.x;
  const int bx   = blockIdx.x;
  const int n0   = blockIdx.y * 128;

  const float* Bw;
  const float* bias;
  int m0 = 0, rowsValid = 128;

  if (GATHER) {
    if (bx >= *numTiles) return;
    const int e    = tileExpert[bx];
    const int p0   = tileP0[bx];
    const int pend = tilePend[bx];
    Bw   = Ball + (long long)e * (long long)K * (long long)ldb;
    bias = biasAll + e * biasStride;
    rowsValid = pend - p0; if (rowsValid > 128) rowsValid = 128;
    if (tid < 128) {
      int idx = tid < rowsValid ? tid : (rowsValid - 1);
      permS[tid] = perm[p0 + idx];
    }
  } else {
    Bw = Ball; bias = biasAll;
    m0 = bx * 128;
  }
  __syncthreads();

  const int wave = tid >> 5, lane = tid & 31;
  const int wm = wave & 3, wn = wave >> 2;        // 4 x 2 wave grid
  const int lm = lane & 15, lh = lane >> 4;

  const bool useTDM = USE_TDM && (ASRC == 1) && !GATHER;

#if USE_TDM
  // Tensor DMA descriptor skeleton for the dense bf16 A tile (128 rows x 32
  // elems, 2B elements, row stride lda, LDS pad 4 DWORDs every 16 DWORDs).
  u32x4 g0 = (u32x4)0u;
  i32x8 g1 = (i32x8)0;
  if (useTDM) {
    unsigned ldsAddr = (unsigned)(unsigned long long)(&As[0]);
    g0.x = 1u;                       // count=1, is_restore=0, gather_mode=0
    g0.y = ldsAddr;                  // lds_addr (bytes)
    // g0.z / g0.w (global_addr + type) are filled per k-step.
    g1.s0 = (1 << 16)                // data_size = 1 (2 bytes)
          | (1 << 20)                // pad_enable
          | (3 << 22)                // pad_interval: every 16 DWORDs (64B row)
          | (3 << 25);               // pad_amount: 4 DWORDs (16B)
    g1.s1 = (int)((unsigned)lda << 16);         // tensor_dim0 = lda (low 16)
    g1.s2 = (int)(((unsigned)B_ROWS & 0xFFFFu) << 16);  // tensor_dim1 low 16
    g1.s3 = (32 << 16);              // tensor_dim1 hi16 = 0, tile_dim0 = 32
    g1.s4 = 128;                     // tile_dim1 = 128 rows, tile_dim2 = 0
    g1.s5 = lda;                     // tensor_dim0_stride (data_size units)
  }
#endif

  v8f acc[2][4];
#pragma unroll
  for (int mt = 0; mt < 2; ++mt)
#pragma unroll
    for (int nt = 0; nt < 4; ++nt) acc[mt][nt] = (v8f)0.0f;

  const int nK = K >> 5;
  for (int kt = 0; kt < nK; ++kt) {
    const int k0 = kt << 5;

    // ---- stage A tile (128 x 32) ----
    if (useTDM) {
#if USE_TDM
      if (wave == 0) {
        unsigned long long ga =
            (unsigned long long)(Abf + (long long)m0 * lda + k0);
        g0.z = (unsigned)(ga & 0xFFFFFFFFu);
        g0.w = (unsigned)((ga >> 32) & 0x01FFFFFFu) | (2u << 30);  // type=2
        __builtin_amdgcn_tensor_load_to_lds(g0, g1, (i32x4)0, (i32x4)0,
                                            (i32x8)0, 0);
      }
#endif
    } else if (ASRC == 0) {
#pragma unroll
      for (int i = 0; i < 4; ++i) {
        int g = i * 256 + tid;            // 1024 float4 groups
        int r = g >> 3;
        int c = (g & 7) << 2;
        f4v v = *(const f4v*)(Af32 + (long long)(m0 + r) * lda + k0 + c);
        u2v pk;
        pk.x = (unsigned)f2bf(v.x) | ((unsigned)f2bf(v.y) << 16);
        pk.y = (unsigned)f2bf(v.z) | ((unsigned)f2bf(v.w) << 16);
        *(u2v*)(&As[r * 40 + c]) = pk;    // one ds_store_b64
      }
    } else {
#pragma unroll
      for (int i = 0; i < 2; ++i) {
        int g = i * 256 + tid;            // 512 16-byte groups
        int r = g >> 2;
        int c = (g & 3) << 3;
        long long row = GATHER ? (long long)permS[r] : (long long)(m0 + r);
        u4v v = *(const u4v*)(Abf + row * lda + k0 + c);
        *(u4v*)(&As[r * 40 + c]) = v;
      }
    }

    // ---- stage B tile (32 x 128), transposed into [n][k] ----
#pragma unroll
    for (int i = 0; i < 4; ++i) {
      int g = i * 256 + tid;              // 1024 float4 groups
      int k = g >> 5;
      int n = (g & 31) << 2;
      f4v v = *(const f4v*)(Bw + (long long)(k0 + k) * ldb + n0 + n);
      Bs[(n + 0) * 40 + k] = f2bf(v.x);
      Bs[(n + 1) * 40 + k] = f2bf(v.y);
      Bs[(n + 2) * 40 + k] = f2bf(v.z);
      Bs[(n + 3) * 40 + k] = f2bf(v.w);
    }

#if USE_TDM
    if (useTDM && wave == 0) __builtin_amdgcn_s_wait_tensorcnt(0);
#endif
    __syncthreads();

    // ---- fragments + WMMA ----
    union ABf { v8us h[2]; v16bf v; };
    ABf a[2], b[4];
#pragma unroll
    for (int mt = 0; mt < 2; ++mt) {
      int mrow = wm * 32 + mt * 16 + lm;
      const unsigned short* p = &As[mrow * 40 + lh * 8];
      a[mt].h[0] = *(const v8us*)p;        // K 0..7  (lh=0) / 8..15  (lh=1)
      a[mt].h[1] = *(const v8us*)(p + 16); // K 16..23 (lh=0) / 24..31 (lh=1)
    }
#pragma unroll
    for (int nt = 0; nt < 4; ++nt) {
      int ncol = wn * 64 + nt * 16 + lm;
      const unsigned short* p = &Bs[ncol * 40 + lh * 8];
      b[nt].h[0] = *(const v8us*)p;        // K lh*8 .. lh*8+7
      b[nt].h[1] = *(const v8us*)(p + 16); // K 16+lh*8 .. 16+lh*8+7
    }
#pragma unroll
    for (int mt = 0; mt < 2; ++mt)
#pragma unroll
      for (int nt = 0; nt < 4; ++nt)
        acc[mt][nt] = __builtin_amdgcn_wmma_f32_16x16x32_bf16(
            false, a[mt].v, false, b[nt].v, (short)0, acc[mt][nt], false, false);
    __syncthreads();
  }

  // ---- epilogue: bias, activation, store (C layout: VGPR r -> M=r+8*lh) ----
#pragma unroll
  for (int nt = 0; nt < 4; ++nt) {
    const int nIdx = n0 + wn * 64 + nt * 16 + lm;
    const float bv = bias[nIdx];
#pragma unroll
    for (int mt = 0; mt < 2; ++mt) {
#pragma unroll
      for (int r = 0; r < 8; ++r) {
        int mloc = wm * 32 + mt * 16 + lh * 8 + r;
        float val = acc[mt][nt][r] + bv;
        if (DO_TANH) val = fast_tanh(val);
        long long row;
        bool ok = true;
        if (GATHER) { ok = (mloc < rowsValid); row = (long long)permS[mloc]; }
        else        { row = (long long)(m0 + mloc); }
        if (ok) {
          if (OUT_BF16) Cbf[row * ldc + nIdx] = f2bf(val);
          else          Cf32[row * ldc + nIdx] = val;
        }
      }
    }
  }
}

// ---------------------------------------------------------------------------
// Host-side launch
// ---------------------------------------------------------------------------
extern "C" void kernel_launch(void* const* d_in, const int* in_sizes, int n_in,
                              void* d_out, int out_size, void* d_ws, size_t ws_size,
                              hipStream_t stream) {
  const float* x    = (const float*)d_in[0];
  const int*   lgts = (const int*)  d_in[1];
  const float* W1   = (const float*)d_in[2];
  const float* b1   = (const float*)d_in[3];
  const float* Wsw  = (const float*)d_in[4];
  const float* bsw  = (const float*)d_in[5];
  const float* W2   = (const float*)d_in[6];
  const float* b2   = (const float*)d_in[7];
  float* out = (float*)d_out;

  char* ws = (char*)d_ws;
  int* sw        = (int*)(ws + 0);             // 4096 ints
  int* counts    = (int*)(ws + 16384);         // 16
  int* offsets   = (int*)(ws + 16448);         // 17
  int* cursor    = (int*)(ws + 16576);         // 16
  int* numTiles  = (int*)(ws + 16640);         // 1
  int* tileExp   = (int*)(ws + 16704);         // 48
  int* tileP0    = (int*)(ws + 16896);         // 48
  int* tilePend  = (int*)(ws + 17088);         // 48
  int* perm      = (int*)(ws + 32768);         // 4096
  unsigned short* h  = (unsigned short*)(ws + 65536);                     // 16 MB bf16
  unsigned short* h2 = (unsigned short*)(ws + 65536 + (size_t)B_ROWS * H_DIM * 2);

  // 1) routing
  init_kernel<<<1, 64, 0, stream>>>(counts, cursor, numTiles);
  router_kernel<<<B_ROWS / 256, 256, 0, stream>>>(lgts, sw, counts);
  scan_kernel<<<1, 32, 0, stream>>>(counts, offsets, cursor,
                                    tileExp, tileP0, tilePend, numTiles);
  scatter_kernel<<<B_ROWS / 256, 256, 0, stream>>>(sw, cursor, perm);

  // 2) h = tanh(x @ W1 + b1)  -> bf16
  gemm_kernel<0, false, true, true><<<dim3(B_ROWS / 128, H_DIM / 128), 256, 0, stream>>>(
      x, nullptr, W1, b1, nullptr, h,
      IN_DIM, IN_DIM, H_DIM, H_DIM,
      nullptr, nullptr, nullptr, nullptr, nullptr, 0);

  // 3) grouped expert GEMM: h2 = tanh(h @ Wsw[e] + bsw[e]) -> bf16
  gemm_kernel<1, true, true, true><<<dim3(MAX_TILES, H_DIM / 128), 256, 0, stream>>>(
      nullptr, h, Wsw, bsw, nullptr, h2,
      H_DIM, H_DIM, H_DIM, H_DIM,
      perm, tileExp, tileP0, tilePend, numTiles, H_DIM);

  // 4) out = h2 @ W2 + b2 -> f32  (A tile staged by the Tensor Data Mover)
  gemm_kernel<1, false, false, false><<<dim3(B_ROWS / 128, OUT_DIM / 128), 256, 0, stream>>>(
      nullptr, h2, W2, b2, out, nullptr,
      H_DIM, H_DIM, OUT_DIM, OUT_DIM,
      nullptr, nullptr, nullptr, nullptr, nullptr, 0);
}
